// GNNActor_75436805587298
// MI455X (gfx1250) — matrix operand order repaired and checked by reading, not compiled, not run on hardware
//
#include <hip/hip_runtime.h>
#include <hip/hip_bf16.h>

// ---------------------------------------------------------------------------
// GNN actor forward for MI455X (gfx1250, wave32, WMMA bf16 w/ fp32 accum)
// Split-K / split-J for wave parallelism; wave-uniform ids via readfirstlane.
// ---------------------------------------------------------------------------

#define N_NODES   4096
#define STATE_DIM 32
#define HIDDEN    64
#define HEADS     4
#define ACTION    8
#define SLOPE     0.01f
#define CONCAT    (HIDDEN * (1 + HEADS))   // 320
#define KSPLIT    8                        // adj-GEMM K chunks (512 each)
#define JSPLIT    4                        // attention j chunks (1024 each)

typedef __bf16 bf16_t;
typedef __attribute__((ext_vector_type(16))) __bf16 v16bf;
typedef __attribute__((ext_vector_type(8)))  __bf16 v8bf;
typedef __attribute__((ext_vector_type(8)))  float  v8f;
typedef __attribute__((ext_vector_type(4)))  float  v4f;

__device__ __forceinline__ float lrelu(float x) { return x > 0.f ? x : SLOPE * x; }

// wave-uniform wave id (forces SGPR -> scalar loop control around WMMA)
__device__ __forceinline__ int uniform_wid(void) {
  int w = __builtin_amdgcn_readfirstlane((int)(threadIdx.x >> 5));
  return (int)blockIdx.x * 8 + w;   // 256 threads/block = 8 waves
}

// ---- WMMA operand loaders --------------------------------------------------
// A (16x32 bf16) layout: lane l -> row m=l&15; K = (e>>3)*16 + ((l>=16)?8:0) + (e&7)
__device__ __forceinline__ v16bf load_a_f32(const float* __restrict__ base, int ld,
                                            int row0, int k0, int lane) {
  int m   = lane & 15;
  int klo = (lane >> 4) * 8;
  const float* p0 = base + (size_t)(row0 + m) * ld + k0 + klo;
  const float* p1 = p0 + 16;
  v4f a0 = *(const v4f*)(p0);
  v4f a1 = *(const v4f*)(p0 + 4);
  v4f b0 = *(const v4f*)(p1);
  v4f b1 = *(const v4f*)(p1 + 4);
  v16bf r;
#pragma unroll
  for (int i = 0; i < 4; ++i) {
    r[i]      = (__bf16)a0[i];
    r[4 + i]  = (__bf16)a1[i];
    r[8 + i]  = (__bf16)b0[i];
    r[12 + i] = (__bf16)b1[i];
  }
  return r;
}

// B (32x16 bf16) layout: lane l -> col n=l&15; K = (e>>3)*16 + ((l>=16)?8:0) + (e&7).
// Source stored col-major bf16 (Bt[col][k]): two 16-byte loads per lane, at k and k+16.
__device__ __forceinline__ v16bf load_b_bf16(const bf16_t* __restrict__ base, int ld,
                                             int col0, int k0, int lane) {
  int n    = lane & 15;
  int koff = (lane >> 4) * 8;
  const bf16_t* p = base + (size_t)(col0 + n) * ld + k0 + koff;
  v8bf lo = *(const v8bf*)(p);
  v8bf hi = *(const v8bf*)(p + 16);
  v16bf r;
#pragma unroll
  for (int i = 0; i < 8; ++i) { r[i] = lo[i]; r[8 + i] = hi[i]; }
  return r;
}

// ---- small prep kernels ----------------------------------------------------

// T1t[o][n] = sum_f state[n][f] * W1[f][o]   (bf16, col-major)
__global__ __launch_bounds__(256) void k_t1(const float* __restrict__ state,
                                            const float* __restrict__ W1,
                                            bf16_t* __restrict__ T1t) {
  int id = blockIdx.x * blockDim.x + threadIdx.x;   // 4096*64
  int n = id & (N_NODES - 1);
  int o = id >> 12;
  float acc = 0.f;
#pragma unroll
  for (int f = 0; f < STATE_DIM; ++f)
    acc = fmaf(state[(size_t)n * STATE_DIM + f], W1[f * HIDDEN + o], acc);
  T1t[(size_t)o * N_NODES + n] = (bf16_t)acc;
}

// T2t[o][n] = sum_f x1[n][f] * W2[f][o]
__global__ __launch_bounds__(256) void k_t2(const float* __restrict__ x1,
                                            const float* __restrict__ W2,
                                            bf16_t* __restrict__ T2t) {
  int id = blockIdx.x * blockDim.x + threadIdx.x;
  int n = id & (N_NODES - 1);
  int o = id >> 12;
  float acc = 0.f;
#pragma unroll
  for (int f = 0; f < HIDDEN; ++f)
    acc = fmaf(x1[(size_t)n * HIDDEN + f], W2[f * HIDDEN + o], acc);
  T2t[(size_t)o * N_NODES + n] = (bf16_t)acc;
}

// Wt[h][o][f] = bf16(W_att[h][f][o])
__global__ __launch_bounds__(256) void k_cvtw(const float* __restrict__ Watt,
                                              bf16_t* __restrict__ Wt) {
  int id = blockIdx.x * blockDim.x + threadIdx.x;   // 4*64*64
  int f = id & 63;
  int o = (id >> 6) & 63;
  int h = id >> 12;
  Wt[id] = (bf16_t)Watt[(size_t)h * HIDDEN * HIDDEN + (size_t)f * HIDDEN + o];
}

// ---- adj GEMM, split-K: one wave per (16x64 strip, K-chunk of 512) ---------
__global__ __launch_bounds__(256) void k_gemm_adj(const float* __restrict__ adj,
                                                  const bf16_t* __restrict__ Tt,
                                                  float* __restrict__ pbuf) {
  int lane = threadIdx.x & 31;
  int wid  = uniform_wid();                 // 0..2047, SGPR
  int strip = wid >> 3;
  int kc    = wid & (KSPLIT - 1);
  int row0  = strip * 16;
  int kbeg  = kc * (N_NODES / KSPLIT);
  int kend  = kbeg + N_NODES / KSPLIT;
  v8f acc[4] = {v8f{}, v8f{}, v8f{}, v8f{}};
  for (int k0 = kbeg; k0 < kend; k0 += 32) {
    v16bf a = load_a_f32(adj, N_NODES, row0, k0, lane);
#pragma unroll
    for (int t = 0; t < 4; ++t) {
      v16bf b = load_b_bf16(Tt, N_NODES, t * 16, k0, lane);
      acc[t] = __builtin_amdgcn_wmma_f32_16x16x32_bf16(false, a, false, b,
                                                       (short)0, acc[t], false, false);
    }
  }
  // partial strip, row-major 16x64: pbuf[(kc*256 + strip)*1024 + r*64 + c]
  float* pb = pbuf + ((size_t)kc * 256 + strip) * (16 * HIDDEN);
  int n  = lane & 15;
  int mb = (lane >> 4) * 8;
#pragma unroll
  for (int t = 0; t < 4; ++t)
#pragma unroll
    for (int v = 0; v < 8; ++v)
      pb[(mb + v) * HIDDEN + t * 16 + n] = acc[t][v];
}

// reduce K-chunks, add bias, relu
__global__ __launch_bounds__(256) void k_gemm_red(const float* __restrict__ pbuf,
                                                  const float* __restrict__ bias,
                                                  float* __restrict__ out, int out_ld) {
  int id  = blockIdx.x * blockDim.x + threadIdx.x;   // 4096*64
  int row = id >> 6;
  int col = id & 63;
  int strip = row >> 4, r = row & 15;
  float s = 0.f;
#pragma unroll
  for (int c = 0; c < KSPLIT; ++c)
    s += pbuf[((size_t)c * 256 + strip) * (16 * HIDDEN) + r * HIDDEN + col];
  out[(size_t)row * out_ld + col] = fmaxf(s + bias[col], 0.f);
}

// ---- per-head h = x2 @ W_att[h]; writes h (fp32) and ht (bf16 transposed) --
__global__ __launch_bounds__(256) void k_heads(const float* __restrict__ xc,
                                               const bf16_t* __restrict__ Wt,
                                               float* __restrict__ h,
                                               bf16_t* __restrict__ ht) {
  int lane = threadIdx.x & 31;
  int wid  = uniform_wid();                 // 0..1023, SGPR
  int head = wid >> 8;
  int row0 = (wid & 255) * 16;
  const bf16_t* W = Wt + (size_t)head * HIDDEN * HIDDEN;
  v8f acc[4] = {v8f{}, v8f{}, v8f{}, v8f{}};
#pragma unroll
  for (int k0 = 0; k0 < HIDDEN; k0 += 32) {
    v16bf a = load_a_f32(xc, CONCAT, row0, k0, lane);   // x2 lives in xc[:, :64]
#pragma unroll
    for (int t = 0; t < 4; ++t) {
      v16bf b = load_b_bf16(W, HIDDEN, t * 16, k0, lane);
      acc[t] = __builtin_amdgcn_wmma_f32_16x16x32_bf16(false, a, false, b,
                                                       (short)0, acc[t], false, false);
    }
  }
  int n  = lane & 15;
  int mb = (lane >> 4) * 8;
#pragma unroll
  for (int t = 0; t < 4; ++t)
#pragma unroll
    for (int v = 0; v < 8; ++v) {
      int row = row0 + mb + v;
      int col = t * 16 + n;
      float y = acc[t][v];
      h[((size_t)head * N_NODES + row) * HIDDEN + col] = y;
      ht[((size_t)head * HIDDEN + col) * N_NODES + row] = (bf16_t)y;
    }
}

// f[h][n] = h[h][n][:] . a1[h],  g[h][n] = h[h][n][:] . a2[h]
__global__ __launch_bounds__(256) void k_fg(const float* __restrict__ h,
                                            const float* __restrict__ a_att,
                                            float* __restrict__ f,
                                            float* __restrict__ g) {
  int id   = blockIdx.x * blockDim.x + threadIdx.x;   // 4*4096
  int n    = id & (N_NODES - 1);
  int head = id >> 12;
  const float* hr = h + ((size_t)head * N_NODES + n) * HIDDEN;
  const float* a1 = a_att + (size_t)head * 2 * HIDDEN;
  const float* a2 = a1 + HIDDEN;
  float af = 0.f, ag = 0.f;
#pragma unroll
  for (int o = 0; o < HIDDEN; ++o) {
    float hv = hr[o];
    af = fmaf(hv, a1[o], af);
    ag = fmaf(hv, a2[o], ag);
  }
  f[id] = af;
  g[id] = ag;
}

__global__ __launch_bounds__(256) void k_gmax(const float* __restrict__ g,
                                              float* __restrict__ gmax) {
  __shared__ float red[256];
  int head = blockIdx.x, t = threadIdx.x;
  float m = -3.4e38f;
  for (int j = t; j < N_NODES; j += 256) m = fmaxf(m, g[(size_t)head * N_NODES + j]);
  red[t] = m;
  __syncthreads();
  for (int s = 128; s > 0; s >>= 1) {
    if (t < s) red[t] = fmaxf(red[t], red[t + s]);
    __syncthreads();
  }
  if (t == 0) gmax[head] = red[0];
}

// ---- flash attention, split-J: wave = (head, strip, j-chunk of 1024) -------
__global__ __launch_bounds__(256) void k_att(const float* __restrict__ f,
                                             const float* __restrict__ g,
                                             const float* __restrict__ gmax,
                                             const bf16_t* __restrict__ ht,
                                             float* __restrict__ patt,
                                             float* __restrict__ psum) {
  int lane = threadIdx.x & 31;
  int wid  = uniform_wid();                 // 0..4095, SGPR
  int head = wid >> 10;
  int rest = wid & 1023;
  int strip = rest >> 2;
  int jc    = rest & (JSPLIT - 1);
  int row0  = strip * 16;
  int m     = lane & 15;
  float fi = f[(size_t)head * N_NODES + row0 + m];
  float mx = lrelu(fi + gmax[head]);   // global row max: lrelu is monotone
  const bf16_t* hb = ht + (size_t)head * HIDDEN * N_NODES;
  const float*  gh = g + (size_t)head * N_NODES;
  v8f acc[4] = {v8f{}, v8f{}, v8f{}, v8f{}};
  float ssum = 0.f;
  int klo  = (lane >> 4) * 8;
  int jbeg = jc * (N_NODES / JSPLIT);
  int jend = jbeg + N_NODES / JSPLIT;
  for (int j0 = jbeg; j0 < jend; j0 += 32) {
    const float* gp = gh + j0 + klo;
    v4f g0 = *(const v4f*)(gp);
    v4f g1 = *(const v4f*)(gp + 4);
    v4f g2 = *(const v4f*)(gp + 16);
    v4f g3 = *(const v4f*)(gp + 20);
    v16bf a;
#pragma unroll
    for (int i = 0; i < 4; ++i) {
      float w0 = __expf(lrelu(fi + g0[i]) - mx);
      float w1 = __expf(lrelu(fi + g1[i]) - mx);
      float w2 = __expf(lrelu(fi + g2[i]) - mx);
      float w3 = __expf(lrelu(fi + g3[i]) - mx);
      ssum += w0 + w1 + w2 + w3;
      a[i]      = (__bf16)w0;
      a[4 + i]  = (__bf16)w1;
      a[8 + i]  = (__bf16)w2;
      a[12 + i] = (__bf16)w3;
    }
#pragma unroll
    for (int t = 0; t < 4; ++t) {
      v16bf b = load_b_bf16(hb, N_NODES, t * 16, j0, lane);
      acc[t] = __builtin_amdgcn_wmma_f32_16x16x32_bf16(false, a, false, b,
                                                       (short)0, acc[t], false, false);
    }
  }
  ssum += __shfl_xor(ssum, 16, 32);   // combine the two K-halves of row m
  size_t base = (size_t)(head * JSPLIT + jc) * 256 + strip;
  float* pb = patt + base * (16 * HIDDEN);
  int n  = lane & 15;
  int mb = (lane >> 4) * 8;
#pragma unroll
  for (int t = 0; t < 4; ++t)
#pragma unroll
    for (int v = 0; v < 8; ++v)
      pb[(mb + v) * HIDDEN + t * 16 + n] = acc[t][v];
  if (lane < 16) psum[base * 16 + lane] = ssum;
}

// reduce j-chunks and normalize into the concat buffer
__global__ __launch_bounds__(256) void k_att_red(const float* __restrict__ patt,
                                                 const float* __restrict__ psum,
                                                 float* __restrict__ xc) {
  int id   = blockIdx.x * blockDim.x + threadIdx.x;   // 4*4096*64
  int head = id >> 18;
  int row  = (id >> 6) & (N_NODES - 1);
  int col  = id & 63;
  int strip = row >> 4, r = row & 15;
  float a = 0.f, s = 0.f;
#pragma unroll
  for (int jc = 0; jc < JSPLIT; ++jc) {
    size_t base = (size_t)(head * JSPLIT + jc) * 256 + strip;
    a += patt[base * (16 * HIDDEN) + r * HIDDEN + col];
    s += psum[base * 16 + r];
  }
  xc[(size_t)row * CONCAT + HIDDEN + head * HIDDEN + col] = a / s;
}

// ---- final MLP: fc1 -> relu -> fc2 -> softmax, one block per row -----------
__global__ __launch_bounds__(64) void k_mlp(const float* __restrict__ xc,
                                            const float* __restrict__ W1,
                                            const float* __restrict__ b1,
                                            const float* __restrict__ W2,
                                            const float* __restrict__ b2,
                                            float* __restrict__ out) {
  __shared__ float row[CONCAT];
  __shared__ float hid[HIDDEN];
  __shared__ float lg[ACTION];
  int r = blockIdx.x, t = threadIdx.x;
  for (int i = t; i < CONCAT; i += 64) row[i] = xc[(size_t)r * CONCAT + i];
  __syncthreads();
  float acc = b1[t];
  for (int q = 0; q < CONCAT; ++q) acc = fmaf(row[q], W1[q * HIDDEN + t], acc);
  hid[t] = fmaxf(acc, 0.f);
  __syncthreads();
  if (t < ACTION) {
    float a = b2[t];
#pragma unroll
    for (int q = 0; q < HIDDEN; ++q) a = fmaf(hid[q], W2[q * ACTION + t], a);
    lg[t] = a;
  }
  __syncthreads();
  if (t == 0) {
    float mx = lg[0];
#pragma unroll
    for (int i = 1; i < ACTION; ++i) mx = fmaxf(mx, lg[i]);
    float e[ACTION], s = 0.f;
#pragma unroll
    for (int i = 0; i < ACTION; ++i) { e[i] = __expf(lg[i] - mx); s += e[i]; }
    float inv = 1.f / s;
#pragma unroll
    for (int i = 0; i < ACTION; ++i) out[(size_t)r * ACTION + i] = e[i] * inv;
  }
}

// ---------------------------------------------------------------------------
extern "C" void kernel_launch(void* const* d_in, const int* in_sizes, int n_in,
                              void* d_out, int out_size, void* d_ws, size_t ws_size,
                              hipStream_t stream) {
  const float* state = (const float*)d_in[0];
  const float* adj   = (const float*)d_in[1];
  const float* W_gc1 = (const float*)d_in[2];
  const float* b_gc1 = (const float*)d_in[3];
  const float* W_gc2 = (const float*)d_in[4];
  const float* b_gc2 = (const float*)d_in[5];
  const float* W_att = (const float*)d_in[6];
  const float* a_att = (const float*)d_in[7];
  const float* W_fc1 = (const float*)d_in[8];
  const float* b_fc1 = (const float*)d_in[9];
  const float* W_fc2 = (const float*)d_in[10];
  const float* b_fc2 = (const float*)d_in[11];
  float* out = (float*)d_out;

  char* ws = (char*)d_ws;
  float*  x1    = (float*)(ws);                        // 1 MB
  bf16_t* T1t   = (bf16_t*)(ws + 0x100000);            // 512 KB
  bf16_t* T2t   = (bf16_t*)(ws + 0x180000);            // 512 KB
  float*  xc    = (float*)(ws + 0x200000);             // 5 MB
  float*  hbuf  = (float*)(ws + 0x700000);             // 4 MB
  bf16_t* ht    = (bf16_t*)(ws + 0xB00000);            // 2 MB
  bf16_t* Wt    = (bf16_t*)(ws + 0xD00000);            // 32 KB
  float*  fb    = (float*)(ws + 0xD10000);             // 64 KB
  float*  gb    = (float*)(ws + 0xD20000);             // 64 KB
  float*  gmax  = (float*)(ws + 0xD30000);             // 16 B
  float*  pgemm = (float*)(ws + 0xD40000);             // 8 MB  (KSPLIT partials)
  float*  patt  = (float*)(ws + 0x1540000);            // 16 MB (JSPLIT partials)
  float*  psum  = (float*)(ws + 0x2540000);            // 256 KB (softmax denoms)

  // layer 1: x1 = relu(adj @ (state@W1) + b1)
  k_t1<<<(N_NODES * HIDDEN) / 256, 256, 0, stream>>>(state, W_gc1, T1t);
  k_gemm_adj<<<256, 256, 0, stream>>>(adj, T1t, pgemm);
  k_gemm_red<<<(N_NODES * HIDDEN) / 256, 256, 0, stream>>>(pgemm, b_gc1, x1, HIDDEN);

  // layer 2: x2 = relu(adj @ (x1@W2) + b2) -> concat buffer cols [0,64)
  k_t2<<<(N_NODES * HIDDEN) / 256, 256, 0, stream>>>(x1, W_gc2, T2t);
  k_gemm_adj<<<256, 256, 0, stream>>>(adj, T2t, pgemm);
  k_gemm_red<<<(N_NODES * HIDDEN) / 256, 256, 0, stream>>>(pgemm, b_gc2, xc, CONCAT);

  // attention heads
  k_cvtw<<<(HEADS * HIDDEN * HIDDEN) / 256, 256, 0, stream>>>(W_att, Wt);
  k_heads<<<128, 256, 0, stream>>>(xc, Wt, hbuf, ht);
  k_fg<<<(HEADS * N_NODES) / 256, 256, 0, stream>>>(hbuf, a_att, fb, gb);
  k_gmax<<<HEADS, 256, 0, stream>>>(gb, gmax);
  k_att<<<512, 256, 0, stream>>>(fb, gb, gmax, ht, patt, psum);
  k_att_red<<<(HEADS * N_NODES * HIDDEN) / 256, 256, 0, stream>>>(patt, psum, xc);

  // output MLP + softmax
  k_mlp<<<N_NODES, 64, 0, stream>>>(xc, W_fc1, b_fc1, W_fc2, b_fc2, out);
}
